// Attention_3375844294750
// MI455X (gfx1250) — compile-verified, matrix operand order checked by reading, not run on hardware
//
#include <hip/hip_runtime.h>
#include <hip/hip_bf16.h>
#include <float.h>

// Problem constants (match the reference)
#define B_     4
#define N_     2048
#define M_     2048
#define QDIM   512
#define CDIM   512
#define HEADS_ 8
#define DHEAD  64
#define INNER_ 512
#define SCALE_ 0.125f   // 64^-0.5

typedef _Float16 v16h __attribute__((ext_vector_type(16)));
typedef _Float16 h8   __attribute__((ext_vector_type(8)));
typedef float    v8f  __attribute__((ext_vector_type(8)));

// Build a 16x32 f16 A/B fragment for v_wmma_f32_16x16x32_f16.
// Per the CDNA5 ISA layout, lane L (row/col = L%16, half = L/16) holds
//   frag[0..7]  = X[row][ 8*half + i]        (contiguous 16B)
//   frag[8..15] = X[row][16 + 8*half + i]    (contiguous 16B)
__device__ __forceinline__ v16h frag_ld(const _Float16* lo_p, const _Float16* hi_p) {
    h8 lo = *(const h8*)lo_p;
    h8 hi = *(const h8*)hi_p;
    v16h r;
#pragma unroll
    for (int i = 0; i < 8; ++i) { r[i] = lo[i]; r[i + 8] = hi[i]; }
    return r;
}

__device__ __forceinline__ v8f wmma16(v16h a, v16h b, v8f c) {
    // (neg_a, A, neg_b, B, c_mod, C, reuse_a, reuse_b)
    return __builtin_amdgcn_wmma_f32_16x16x32_f16(false, a, false, b, (short)0, c,
                                                  false, false);
}

// ---------------------------------------------------------------------------
// f32 -> f16 elementwise convert
__global__ void cvt_f32_to_h(const float* __restrict__ src,
                             _Float16* __restrict__ dst, int n) {
    int i = blockIdx.x * blockDim.x + threadIdx.x;
    if (i < n) dst[i] = (_Float16)src[i];
}

// Transpose-convert weight W[in=rows][... ld ...] (taking cols [col0, col0+cols))
// into Wt[out][in] f16, so GEMM B-operand columns become contiguous rows.
__global__ void cvt_transpose(const float* __restrict__ src,
                              _Float16* __restrict__ dst,
                              int rows, int cols, int ld, int col0) {
    int i = blockIdx.x * blockDim.x + threadIdx.x;
    if (i >= rows * cols) return;
    int o = i % cols;
    int r = i / cols;
    dst[(size_t)o * rows + r] = (_Float16)src[(size_t)r * ld + col0 + o];
}

// ---------------------------------------------------------------------------
// f16 WMMA GEMM, register-blocked: each wave computes a 16x64 strip
// (4 accumulators) so the A fragment is loaded once and reused 4x per k-step.
// C[m][n] = sum_k A[m][k] * Bt[n][k]
// mode: 0 = f16 row-major store,
//       1 = f16 store transposed per batch -> out[b][n][m%Mdim] (for V),
//       2 = f32 store + bias (final projection into d_out).
__global__ void gemm_wmma(const _Float16* __restrict__ A,
                          const _Float16* __restrict__ Bt,
                          int Ncols, int K,
                          _Float16* __restrict__ outH,
                          float* __restrict__ outF,
                          const float* __restrict__ bias,
                          int mode, int Mdim) {
    const int lane = threadIdx.x & 31;
    const int col  = lane & 15;
    const int half = lane >> 4;
    const int kb0  = 8 * half;
    const int kb1  = 16 + 8 * half;
    const int n0   = blockIdx.x * 64;   // 4 x 16-column tiles
    const int m0   = blockIdx.y * 16;

    const _Float16* arow = A + (size_t)(m0 + col) * K;
    const _Float16* brow[4];
#pragma unroll
    for (int t = 0; t < 4; ++t)
        brow[t] = Bt + (size_t)(n0 + t * 16 + col) * K;

    v8f acc[4];
#pragma unroll
    for (int t = 0; t < 4; ++t) acc[t] = (v8f){};

    for (int k0 = 0; k0 < K; k0 += 32) {
        const v16h a = frag_ld(arow + k0 + kb0, arow + k0 + kb1);
#pragma unroll
        for (int t = 0; t < 4; ++t)
            acc[t] = wmma16(a, frag_ld(brow[t] + k0 + kb0, brow[t] + k0 + kb1),
                            acc[t]);
    }

    // C-layout: lane (col, half), VGPR r  <->  row = 8*half + r, column = col
    const int mrow0 = m0 + 8 * half;
    if (mode == 0) {
#pragma unroll
        for (int t = 0; t < 4; ++t)
#pragma unroll
            for (int r = 0; r < 8; ++r)
                outH[(size_t)(mrow0 + r) * Ncols + n0 + t * 16 + col] =
                    (_Float16)acc[t][r];
    } else if (mode == 1) {
        // transposed store: consecutive r -> consecutive m -> packed 16B stores
        const int bidx = mrow0 / Mdim;
        const int mloc = mrow0 % Mdim;
#pragma unroll
        for (int t = 0; t < 4; ++t) {
            h8 pk = {};
#pragma unroll
            for (int r = 0; r < 8; ++r) pk[r] = (_Float16)acc[t][r];
            *(h8*)(outH + ((size_t)bidx * Ncols + n0 + t * 16 + col) * Mdim + mloc) = pk;
        }
    } else {
#pragma unroll
        for (int t = 0; t < 4; ++t) {
            const float bv = bias[n0 + t * 16 + col];
#pragma unroll
            for (int r = 0; r < 8; ++r)
                outF[(size_t)(mrow0 + r) * Ncols + n0 + t * 16 + col] =
                    acc[t][r] + bv;
        }
    }
}

// ---------------------------------------------------------------------------
// Flash attention: one wave handles 16 query rows x one head, streaming keys
// in tiles of 32 with online softmax. Q: [B*N][INNER], K: [B*M][INNER],
// Vt: [B][INNER][M] (d-major), O: [B*N][INNER] f16.
__global__ void flash_attn(const _Float16* __restrict__ Q,
                           const _Float16* __restrict__ Kh,
                           const _Float16* __restrict__ Vt,
                           const int* __restrict__ mask,
                           _Float16* __restrict__ O) {
    const int lane = threadIdx.x & 31;
    const int col  = lane & 15;
    const int half = lane >> 4;
    const int kb0  = 8 * half;
    const int kb1  = 16 + 8 * half;
    const int bh   = blockIdx.x;
    const int b    = bh / HEADS_;
    const int h    = bh % HEADS_;
    const int q0   = blockIdx.y * 16;

    __shared__ _Float16 sm[16 * 32];   // P tile staging (C-layout -> A-layout)

    // Q fragments stay resident: two 16x32 chunks covering d = 0..63
    const _Float16* qrow = Q + (size_t)(b * N_ + q0 + col) * INNER_ + h * DHEAD;
    const v16h aq0 = frag_ld(qrow + kb0,      qrow + kb1);
    const v16h aq1 = frag_ld(qrow + 32 + kb0, qrow + 32 + kb1);

    const _Float16* kbase = Kh + (size_t)b * M_ * INNER_ + h * DHEAD;
    const _Float16* vbase = Vt + ((size_t)b * INNER_ + h * DHEAD) * M_;
    const int* mrow = mask + b * M_;

    float rm[8], rl[8];
    v8f acc[4];
#pragma unroll
    for (int r = 0; r < 8; ++r) { rm[r] = -FLT_MAX; rl[r] = 0.0f; }
#pragma unroll
    for (int g = 0; g < 4; ++g) acc[g] = (v8f){};

    for (int j0 = 0; j0 < M_; j0 += 32) {
        // ---- S = Q*K^T for 32 keys: two 16-column C tiles, 4 WMMAs total
        const _Float16* kr0 = kbase + (size_t)(j0 + col) * INNER_;
        const _Float16* kr1 = kbase + (size_t)(j0 + 16 + col) * INNER_;
        v8f s0 = {}, s1 = {};
        s0 = wmma16(aq0, frag_ld(kr0 + kb0,      kr0 + kb1),      s0);
        s0 = wmma16(aq1, frag_ld(kr0 + 32 + kb0, kr0 + 32 + kb1), s0);
        s1 = wmma16(aq0, frag_ld(kr1 + kb0,      kr1 + kb1),      s1);
        s1 = wmma16(aq1, frag_ld(kr1 + 32 + kb0, kr1 + 32 + kb1), s1);

        // Prefetch next tile's K rows (lowers to global_prefetch_b8) while the
        // softmax VALU work below hides the latency.
        if (j0 + 32 < M_) {
            __builtin_prefetch(kbase + (size_t)(j0 + 32 + col) * INNER_, 0, 1);
            __builtin_prefetch(kbase + (size_t)(j0 + 48 + col) * INNER_, 0, 1);
        }

        const bool mk0 = mrow[j0 + col] != 0;
        const bool mk1 = mrow[j0 + 16 + col] != 0;

        // ---- online softmax; row (8*half + r) lives on the 16 lanes of
        // this half, so row reductions are xor-shuffles of width 16.
        float corr[8];
#pragma unroll
        for (int r = 0; r < 8; ++r) {
            float a0 = mk0 ? s0[r] * SCALE_ : -FLT_MAX;
            float a1 = mk1 ? s1[r] * SCALE_ : -FLT_MAX;
            float t = fmaxf(a0, a1);
            for (int msk = 1; msk < 16; msk <<= 1)
                t = fmaxf(t, __shfl_xor(t, msk, 16));
            const float mn = fmaxf(rm[r], t);
            corr[r] = __expf(rm[r] - mn);
            rm[r] = mn;
            const float p0 = __expf(a0 - mn);
            const float p1 = __expf(a1 - mn);
            sm[(8 * half + r) * 32 + col]      = (_Float16)p0;
            sm[(8 * half + r) * 32 + 16 + col] = (_Float16)p1;
            float ps = p0 + p1;
            for (int msk = 1; msk < 16; msk <<= 1)
                ps += __shfl_xor(ps, msk, 16);
            rl[r] = rl[r] * corr[r] + ps;
        }
#pragma unroll
        for (int g = 0; g < 4; ++g)
#pragma unroll
            for (int r = 0; r < 8; ++r) acc[g][r] *= corr[r];
        __syncthreads();

        // ---- reload P (16x32) in A-fragment layout from LDS (ds_load_b128)
        const _Float16* prow = sm + col * 32;
        const v16h ap = frag_ld(prow + kb0, prow + kb1);

        // ---- O += P @ V : V is d-major so B-fragment loads are contiguous
#pragma unroll
        for (int g = 0; g < 4; ++g) {
            const _Float16* vrow = vbase + (size_t)(g * 16 + col) * M_ + j0;
            acc[g] = wmma16(ap, frag_ld(vrow + kb0, vrow + kb1), acc[g]);
        }
        __syncthreads();
    }

    float inv[8];
#pragma unroll
    for (int r = 0; r < 8; ++r) inv[r] = 1.0f / rl[r];
#pragma unroll
    for (int g = 0; g < 4; ++g)
#pragma unroll
        for (int r = 0; r < 8; ++r)
            O[(size_t)(b * N_ + q0 + 8 * half + r) * INNER_ + h * DHEAD + g * 16 + col] =
                (_Float16)(acc[g][r] * inv[r]);
}

// ---------------------------------------------------------------------------
extern "C" void kernel_launch(void* const* d_in, const int* in_sizes, int n_in,
                              void* d_out, int out_size, void* d_ws, size_t ws_size,
                              hipStream_t stream) {
    const float* x    = (const float*)d_in[0];
    const float* ctx  = (const float*)d_in[1];
    const int*   mask = (const int*)d_in[2];
    const float* Wq   = (const float*)d_in[3];
    const float* Wkv  = (const float*)d_in[4];
    const float* Wo   = (const float*)d_in[5];
    const float* bo   = (const float*)d_in[6];
    float* out = (float*)d_out;

    // Workspace layout (f16): six 8192x512 tensors + four 512x512 weights ~ 50 MB
    _Float16* W = (_Float16*)d_ws;
    const size_t SZ = (size_t)B_ * N_ * INNER_;   // 4M elements
    _Float16* x_h  = W;
    _Float16* c_h  = W + SZ;
    _Float16* q_h  = W + 2 * SZ;
    _Float16* k_h  = W + 3 * SZ;
    _Float16* vt_h = W + 4 * SZ;   // [B][INNER][M] (d-major V)
    _Float16* ao_h = W + 5 * SZ;
    _Float16* WqT  = W + 6 * SZ;
    _Float16* WkT  = WqT + 512 * 512;
    _Float16* WvT  = WkT + 512 * 512;
    _Float16* WoT  = WvT + 512 * 512;

    const int nE = (int)SZ;
    cvt_f32_to_h<<<(nE + 255) / 256, 256, 0, stream>>>(x,   x_h, nE);
    cvt_f32_to_h<<<(nE + 255) / 256, 256, 0, stream>>>(ctx, c_h, nE);

    const int nW = 512 * 512;
    cvt_transpose<<<(nW + 255) / 256, 256, 0, stream>>>(Wq,  WqT, 512, 512, 512,  0);
    cvt_transpose<<<(nW + 255) / 256, 256, 0, stream>>>(Wkv, WkT, 512, 512, 1024, 0);
    cvt_transpose<<<(nW + 255) / 256, 256, 0, stream>>>(Wkv, WvT, 512, 512, 1024, 512);
    cvt_transpose<<<(nW + 255) / 256, 256, 0, stream>>>(Wo,  WoT, 512, 512, 512,  0);

    // 16x64 output strip per wave: grid (Ncols/64, Mrows/16)
    dim3 gg(INNER_ / 64, (B_ * N_) / 16);   // 8 x 512
    gemm_wmma<<<gg, 32, 0, stream>>>(x_h, WqT, INNER_, QDIM, q_h,  nullptr, nullptr, 0, 0);
    gemm_wmma<<<gg, 32, 0, stream>>>(c_h, WkT, INNER_, CDIM, k_h,  nullptr, nullptr, 0, 0);
    gemm_wmma<<<gg, 32, 0, stream>>>(c_h, WvT, INNER_, CDIM, vt_h, nullptr, nullptr, 1, M_);

    flash_attn<<<dim3(B_ * HEADS_, N_ / 16), 32, 0, stream>>>(q_h, k_h, vt_h, mask, ao_h);

    gemm_wmma<<<dim3(QDIM / 64, (B_ * N_) / 16), 32, 0, stream>>>(
        ao_h, WoT, QDIM, INNER_, nullptr, out, bo, 2, 0);
}
